// Policy_27410481283588
// MI455X (gfx1250) — compile-verified, matrix operand order checked by reading, not run on hardware
//
#include <hip/hip_runtime.h>
#include <cstddef>

// ---------------------------------------------------------------------------
// MI455X / gfx1250 implementation.
//  - All large GEMMs run through v_wmma_f32_16x16x32_f16 (f16 in, f32 acc).
//  - LSTM recurrence runs in ONE workgroup with h (f16) + c (f32) resident in
//    the 320KB WGP LDS for all 512 steps; R streams from L2 each step.
//  - wx = LN(h2 @ Wx^T)*gx is hoisted out of the scan (T-parallel) as a bulk
//    WMMA GEMM + LN pass.
//  - k-loops are partially unrolled (4/8) so fragment live ranges stay small:
//    full unroll made LICM hoist 128 VGPRs of loop-invariant A fragments in
//    the scan kernel and spill them to scratch inside the recurrence.
// ---------------------------------------------------------------------------

typedef __attribute__((ext_vector_type(16))) _Float16 v16h;
typedef __attribute__((ext_vector_type(8)))  float    v8f;

#define TT   512
#define BB   64
#define OBSD 128
#define ADIM 18
#define GOALD 16
#define HD   512
#define ED   256
#define HALFD 128
#define LTOK 5
#define NTOK (TT*BB)           // 32768
#define L5N  ((size_t)LTOK*NTOK)

__device__ __forceinline__ float sigf(float x) { return 1.f / (1.f + __expf(-x)); }

// ---------------------------------------------------------------------------
// Generic f32 -> f16 conversion (weights + observations)
// ---------------------------------------------------------------------------
__global__ void cvt_f32_f16(const float* __restrict__ s, _Float16* __restrict__ d, long n) {
    long i = (long)blockIdx.x * blockDim.x + threadIdx.x;
    if (i < n) d[i] = (_Float16)s[i];
}

// Pad Wp (18,512) -> (32,512) f16 with zero rows so the logits GEMM tiles evenly.
__global__ void pad_wp(const float* __restrict__ Wp, _Float16* __restrict__ dst) {
    int i = blockIdx.x * 256 + threadIdx.x;
    if (i >= 32 * HD) return;
    int row = i >> 9;
    dst[i] = (row < ADIM) ? (_Float16)Wp[i] : (_Float16)0.f;
}

// ---------------------------------------------------------------------------
// WMMA GEMM:  out[n][m] = act( sum_k A[n][k] * W[m][k] + bias[m] (+ resid) )
// A: (rows, K) f16 row-major.  W: (M, K) f16 row-major (i.e. x @ W^T).
// One wave per 16x16 output tile; unroll 8 -> bounded fragment live ranges.
// ACT: 0 none, 1 tanh, 2 relu
// ---------------------------------------------------------------------------
template<int K, int ACT>
__global__ __launch_bounds__(32)
void gemm_wmma(const _Float16* __restrict__ A, const _Float16* __restrict__ W,
               const float* __restrict__ bias, const _Float16* __restrict__ resid,
               _Float16* __restrict__ outH, float* __restrict__ outF,
               int ldo, int Mvalid)
{
    const int lane = threadIdx.x;
    const int row0 = blockIdx.x * 16;
    const int col0 = blockIdx.y * 16;
    const int r  = lane & 15;
    const int kh = (lane >> 4) << 4;        // lanes 16..31 hold K+16..K+31
    const _Float16* Ap = A + (size_t)(row0 + r) * K + kh;
    const _Float16* Wp = W + (size_t)(col0 + r) * K + kh;

    v8f acc = {};
#pragma unroll 8
    for (int k0 = 0; k0 < K; k0 += 32) {
        v16h a = *(const v16h*)(Ap + k0);
        v16h b = *(const v16h*)(Wp + k0);
        acc = __builtin_amdgcn_wmma_f32_16x16x32_f16(false, a, false, b,
                                                     (short)0, acc, false, false);
    }

    const int ocol = col0 + (lane & 15);
    const int orow = row0 + ((lane >> 4) << 3);   // VGPR i -> row orow+i
    const bool colok = (ocol < Mvalid);
    float bv = (bias && colok) ? bias[ocol] : 0.f;
#pragma unroll
    for (int i = 0; i < 8; ++i) {
        if (!colok) continue;
        size_t idx = (size_t)(orow + i) * ldo + ocol;
        float v = acc[i] + bv;
        if (resid) v += (float)resid[idx];
        if (ACT == 1) v = tanhf(v);
        else if (ACT == 2) v = fmaxf(v, 0.f);
        if (outH) outH[idx] = (_Float16)v;
        if (outF) outF[idx] = v;
    }
}

// ---------------------------------------------------------------------------
// Command embeddings -> x (L, N, E) f16.  L order: goal, action, horizon,
// reward, ret (reference stacking order).
// ---------------------------------------------------------------------------
__global__ void embed_kernel(const float* __restrict__ goal, const int* __restrict__ act,
                             const float* __restrict__ hor, const float* __restrict__ rew,
                             const float* __restrict__ ret,
                             const float* Wg, const float* bg, const float* eg,
                             const float* Wa, const float* ba, const float* ea,
                             const float* Wh, const float* bh, const float* eh,
                             const float* Wr, const float* br, const float* er,
                             const float* Wt, const float* bt, const float* et,
                             _Float16* __restrict__ x)
{
    int idx = blockIdx.x * blockDim.x + threadIdx.x;     // over NTOK*HALFD
    if (idx >= NTOK * HALFD) return;
    int n = idx >> 7, j = idx & (HALFD - 1);

    // l=0 goal
    {
        float s = bg[j];
#pragma unroll
        for (int d = 0; d < GOALD; ++d) s += goal[(size_t)n * GOALD + d] * Wg[j * GOALD + d];
        _Float16* p = x + ((size_t)0 * NTOK + n) * ED;
        p[j] = (_Float16)s; p[HALFD + j] = (_Float16)eg[j];
    }
    // l=1 action (one-hot -> column gather)
    {
        int a = act[n];
        _Float16* p = x + ((size_t)1 * NTOK + n) * ED;
        p[j] = (_Float16)(Wa[j * ADIM + a] + ba[j]); p[HALFD + j] = (_Float16)ea[j];
    }
    // l=2 horizon, l=3 reward, l=4 ret  (d=1 embeddings)
    {
        _Float16* p = x + ((size_t)2 * NTOK + n) * ED;
        p[j] = (_Float16)(Wh[j] * hor[n] + bh[j]); p[HALFD + j] = (_Float16)eh[j];
    }
    {
        _Float16* p = x + ((size_t)3 * NTOK + n) * ED;
        p[j] = (_Float16)(Wr[j] * rew[n] + br[j]); p[HALFD + j] = (_Float16)er[j];
    }
    {
        _Float16* p = x + ((size_t)4 * NTOK + n) * ED;
        p[j] = (_Float16)(Wt[j] * ret[n] + bt[j]); p[HALFD + j] = (_Float16)et[j];
    }
}

// ---------------------------------------------------------------------------
// LayerNorm over rows of width D (wave per row, wave32 shuffles).
// ---------------------------------------------------------------------------
__global__ __launch_bounds__(256)
void ln_rows(const _Float16* __restrict__ x, const float* __restrict__ g,
             const float* __restrict__ b, _Float16* __restrict__ outH,
             float* __restrict__ outF, int D, long rows)
{
    long row = (long)blockIdx.x * 8 + (threadIdx.x >> 5);
    if (row >= rows) return;
    int lane = threadIdx.x & 31;
    const _Float16* xp = x + row * D;
    float s = 0.f, q = 0.f;
    for (int d = lane; d < D; d += 32) { float v = (float)xp[d]; s += v; q += v * v; }
#pragma unroll
    for (int m = 16; m; m >>= 1) { s += __shfl_xor(s, m, 32); q += __shfl_xor(q, m, 32); }
    float mu = s / D, var = q / D - mu * mu, rs = rsqrtf(var + 1e-5f);
    for (int d = lane; d < D; d += 32) {
        float v = ((float)xp[d] - mu) * rs * g[d] + (b ? b[d] : 0.f);
        if (outH) outH[row * D + d] = (_Float16)v;
        if (outF) outF[row * D + d] = v;
    }
}

// ---------------------------------------------------------------------------
// Attention over L=5 commands, one wave per token; q/k/v live in registers.
// qkv rows are (l*NTOK + n) x 768 (q | k | v).
// ---------------------------------------------------------------------------
__global__ __launch_bounds__(256)
void attn_kernel(const _Float16* __restrict__ qkv, _Float16* __restrict__ obuf, int Ntok)
{
    int token = blockIdx.x * 8 + (threadIdx.x >> 5);
    if (token >= Ntok) return;
    int lane = threadIdx.x & 31;
    float q[5][8], k[5][8], v[5][8];
#pragma unroll
    for (int l = 0; l < 5; ++l) {
        const _Float16* p = qkv + ((size_t)l * Ntok + token) * 768 + lane * 8;
#pragma unroll
        for (int i = 0; i < 8; ++i) {
            q[l][i] = (float)p[i]; k[l][i] = (float)p[256 + i]; v[l][i] = (float)p[512 + i];
        }
    }
    float s[5][5];
#pragma unroll
    for (int l = 0; l < 5; ++l)
#pragma unroll
        for (int m = 0; m < 5; ++m) {
            float d = 0.f;
#pragma unroll
            for (int i = 0; i < 8; ++i) d += q[l][i] * k[m][i];
#pragma unroll
            for (int mm = 16; mm; mm >>= 1) d += __shfl_xor(d, mm, 32);
            s[l][m] = d * 0.0625f;                      // 1/sqrt(256)
        }
#pragma unroll
    for (int l = 0; l < 5; ++l) {
        float mx = s[l][0];
#pragma unroll
        for (int m = 1; m < 5; ++m) mx = fmaxf(mx, s[l][m]);
        float den = 0.f;
#pragma unroll
        for (int m = 0; m < 5; ++m) { s[l][m] = __expf(s[l][m] - mx); den += s[l][m]; }
        float inv = 1.f / den;
#pragma unroll
        for (int m = 0; m < 5; ++m) s[l][m] *= inv;
    }
#pragma unroll
    for (int l = 0; l < 5; ++l) {
        _Float16* op = obuf + ((size_t)l * Ntok + token) * ED + lane * 8;
#pragma unroll
        for (int i = 0; i < 8; ++i) {
            float o = 0.f;
#pragma unroll
            for (int m = 0; m < 5; ++m) o += s[l][m] * v[m][i];
            op[i] = (_Float16)o;
        }
    }
}

// max over L commands -> emb (N, E) f16
__global__ void maxL_kernel(const _Float16* __restrict__ x, _Float16* __restrict__ emb, int Ntok)
{
    int idx = blockIdx.x * blockDim.x + threadIdx.x;     // over Ntok*ED
    if (idx >= Ntok * ED) return;
    int n = idx >> 8, e = idx & (ED - 1);
    float m = -3.4e38f;
#pragma unroll
    for (int l = 0; l < 5; ++l)
        m = fmaxf(m, (float)x[((size_t)l * Ntok + n) * ED + e]);
    emb[(size_t)n * ED + e] = (_Float16)m;
}

// h2 = ga * sigmoid(gb)
__global__ void gate_combine(const _Float16* __restrict__ ga, const _Float16* __restrict__ gb,
                             _Float16* __restrict__ h2, long n)
{
    long i = (long)blockIdx.x * blockDim.x + threadIdx.x;
    if (i < n) h2[i] = (_Float16)((float)ga[i] * sigf((float)gb[i]));
}

// ---------------------------------------------------------------------------
// LayerNormLSTM scan: ONE workgroup (32 waves), h (f16) + c (f32) resident in
// LDS (~206 KB of the 320 KB WGP pool).  Each step: 512 WMMA tiles of
// rh = h @ R^T, LN(2048) via LDS reduction, gates, LN(512) on c, h update.
// k-loop unrolled by 4 only: A fragments are tile-loop-invariant, and full
// unroll caused 128 VGPRs of hoisted fragments to spill to scratch.
// ---------------------------------------------------------------------------
__global__ __launch_bounds__(1024)
void lstm_scan(const _Float16* __restrict__ Rh,    // (2048,512) f16
               const float* __restrict__ wx,       // (T,64,2048) f32 (pre-LN'd, *gx)
               const float* __restrict__ gr, const float* __restrict__ blstm,
               const float* __restrict__ gc, const float* __restrict__ bc,
               const float* __restrict__ h0, const float* __restrict__ c0,
               float* __restrict__ vg_raw,         // (64,2048) f32 scratch (global)
               _Float16* __restrict__ h3,          // (T,64,512) f16
               float* __restrict__ hn, float* __restrict__ cn, int Tsteps)
{
    extern __shared__ char smem[];
    float*    cst  = (float*)smem;                                    // 64*512 f32
    _Float16* hh   = (_Float16*)(smem + (size_t)BB * HD * 4);         // 64*512 f16
    float*    redS = (float*)(smem + (size_t)BB * HD * 4 + (size_t)BB * HD * 2);
    float*    redQ = redS + BB * 16;
    float*    muA  = redQ + BB * 16;
    float*    rsA  = muA + BB;
    float*    mcA  = rsA + BB;
    float*    rcA  = mcA + BB;

    const int tid = threadIdx.x, lane = tid & 31, wv = tid >> 5;

    for (int idx = tid; idx < BB * HD; idx += 1024) {
        int j = idx & (HD - 1);
        cst[idx] = c0[j];
        hh[idx]  = (_Float16)h0[j];
    }
    __syncthreads();

    const int r  = lane & 15;
    const int kh = (lane >> 4) << 4;

    for (int t = 0; t < Tsteps; ++t) {
        // --- rh = h @ R^T : 4 x 128 tiles of 16x16, K=512 ---
        for (int tile = wv; tile < 512; tile += 32) {
            int rt = tile & 3, ct = tile >> 2;
            const _Float16* Ap = hh + (size_t)(rt * 16 + r) * HD + kh;
            const _Float16* Bp = Rh + (size_t)(ct * 16 + r) * HD + kh;
            v8f acc = {};
#pragma unroll 4
            for (int k0 = 0; k0 < HD; k0 += 32) {
                v16h a = *(const v16h*)(Ap + k0);
                v16h b = *(const v16h*)(Bp + k0);
                acc = __builtin_amdgcn_wmma_f32_16x16x32_f16(false, a, false, b,
                                                             (short)0, acc, false, false);
            }
            int orow = rt * 16 + ((lane >> 4) << 3);
            int ocol = ct * 16 + (lane & 15);
#pragma unroll
            for (int i = 0; i < 8; ++i)
                vg_raw[(size_t)(orow + i) * 2048 + ocol] = acc[i];
        }
        __threadfence();
        __syncthreads();

        // --- LN stats over 2048 per row (16 threads/row, 128 elems each) ---
        {
            int row = tid >> 4, seg = tid & 15;
            const float* p = vg_raw + (size_t)row * 2048 + seg * 128;
            float s = 0.f, q = 0.f;
            for (int i = 0; i < 128; ++i) { float v = p[i]; s += v; q += v * v; }
            redS[row * 16 + seg] = s; redQ[row * 16 + seg] = q;
        }
        __syncthreads();
        if ((tid & 15) == 0) {
            int row = tid >> 4;
            float s = 0.f, q = 0.f;
            for (int i = 0; i < 16; ++i) { s += redS[row * 16 + i]; q += redQ[row * 16 + i]; }
            float mu = s / 2048.f, var = q / 2048.f - mu * mu;
            muA[row] = mu; rsA[row] = rsqrtf(var + 1e-5f);
        }
        __syncthreads();

        // --- gates i,g,f -> c update ---
        const float* wxt = wx + (size_t)t * BB * 2048;
        for (int idx = tid; idx < BB * HD; idx += 1024) {
            int n = idx >> 9, j = idx & (HD - 1);
            float mu = muA[n], rs = rsA[n];
            const float* vr  = vg_raw + (size_t)n * 2048;
            const float* wxp = wxt    + (size_t)n * 2048;
            float iv = wxp[j]          + (vr[j]          - mu) * rs * gr[j]          + blstm[j];
            float gv = wxp[512 + j]    + (vr[512 + j]    - mu) * rs * gr[512 + j]    + blstm[512 + j];
            float fv = wxp[1024 + j]   + (vr[1024 + j]   - mu) * rs * gr[1024 + j]   + blstm[1024 + j];
            cst[idx] = sigf(fv) * cst[idx] + sigf(iv) * tanhf(gv);
        }
        __syncthreads();

        // --- LN stats over 512 per row of c ---
        {
            int row = tid >> 4, seg = tid & 15;
            const float* p = cst + row * HD + seg * 32;
            float s = 0.f, q = 0.f;
            for (int i = 0; i < 32; ++i) { float v = p[i]; s += v; q += v * v; }
            redS[row * 16 + seg] = s; redQ[row * 16 + seg] = q;
        }
        __syncthreads();
        if ((tid & 15) == 0) {
            int row = tid >> 4;
            float s = 0.f, q = 0.f;
            for (int i = 0; i < 16; ++i) { s += redS[row * 16 + i]; q += redQ[row * 16 + i]; }
            float mu = s / 512.f, var = q / 512.f - mu * mu;
            mcA[row] = mu; rcA[row] = rsqrtf(var + 1e-5f);
        }
        __syncthreads();

        // --- o gate, h = sig(o)*tanh(LN(c)) ---
        for (int idx = tid; idx < BB * HD; idx += 1024) {
            int n = idx >> 9, j = idx & (HD - 1);
            float mu = muA[n], rs = rsA[n];
            const float* vr  = vg_raw + (size_t)n * 2048;
            const float* wxp = wxt    + (size_t)n * 2048;
            float ov = wxp[1536 + j] + (vr[1536 + j] - mu) * rs * gr[1536 + j] + blstm[1536 + j];
            float cl = (cst[idx] - mcA[n]) * rcA[n] * gc[j] + bc[j];
            float hv = sigf(ov) * tanhf(cl);
            hh[idx] = (_Float16)hv;
            h3[((size_t)t * BB + n) * HD + j] = (_Float16)hv;
            if (t == Tsteps - 1) { hn[idx] = hv; cn[idx] = cst[idx]; }
        }
        __syncthreads();
    }
}

// ---------------------------------------------------------------------------
// Host orchestration
// ---------------------------------------------------------------------------
extern "C" void kernel_launch(void* const* d_in, const int* in_sizes, int n_in,
                              void* d_out, int out_size, void* d_ws, size_t ws_size,
                              hipStream_t stream)
{
    (void)in_sizes; (void)n_in; (void)out_size; (void)ws_size;
    // ---- inputs (setup_inputs() dict order) ----
    const float* obs  = (const float*)d_in[0];
    const float* goal = (const float*)d_in[1];
    const int*   pact = (const int*)d_in[2];
    const float* hor  = (const float*)d_in[3];
    const float* rew  = (const float*)d_in[4];
    const float* retn = (const float*)d_in[5];
    // ---- params: JAX tree-flatten (sorted keys, ASCII; uppercase first) ----
    const float* R     = (const float*)d_in[6];    // (2048,512)
    const float* W1    = (const float*)d_in[7];    // (512,128)
    const float* W2    = (const float*)d_in[8];    // (512,512)
    const float* Wcg   = (const float*)d_in[9];    // (512,256)
    const float* Wf1   = (const float*)d_in[10];   // (512,256)
    const float* Wf2   = (const float*)d_in[11];   // (256,512)
    const float* Win   = (const float*)d_in[12];   // (768,256)
    const float* Wo    = (const float*)d_in[13];   // (256,256)
    const float* Wp    = (const float*)d_in[14];   // (18,512)
    const float* Wx    = (const float*)d_in[15];   // (2048,512)
    const float* b1    = (const float*)d_in[16];
    const float* b2    = (const float*)d_in[17];
    const float* bc    = (const float*)d_in[18];
    const float* bcg   = (const float*)d_in[19];
    const float* be1   = (const float*)d_in[20];
    const float* be2   = (const float*)d_in[21];
    const float* bf1   = (const float*)d_in[22];
    const float* bf2   = (const float*)d_in[23];
    const float* bin   = (const float*)d_in[24];
    const float* blstm = (const float*)d_in[25];
    const float* bo    = (const float*)d_in[26];
    const float* bp    = (const float*)d_in[27];
    const float* c0    = (const float*)d_in[28];
    // ce: action, goal, horizon, ret, reward  (each: W, b, enc)
    const float* ceaW = (const float*)d_in[29]; const float* ceab = (const float*)d_in[30]; const float* ceae = (const float*)d_in[31];
    const float* cegW = (const float*)d_in[32]; const float* cegb = (const float*)d_in[33]; const float* cege = (const float*)d_in[34];
    const float* cehW = (const float*)d_in[35]; const float* cehb = (const float*)d_in[36]; const float* cehe = (const float*)d_in[37];
    const float* cetW = (const float*)d_in[38]; const float* cetb = (const float*)d_in[39]; const float* cete = (const float*)d_in[40]; // ret
    const float* cerW = (const float*)d_in[41]; const float* cerb = (const float*)d_in[42]; const float* cere = (const float*)d_in[43]; // reward
    const float* g1 = (const float*)d_in[44];
    const float* g2 = (const float*)d_in[45];
    const float* gc = (const float*)d_in[46];
    const float* gr = (const float*)d_in[47];
    const float* gx = (const float*)d_in[48];
    const float* h0 = (const float*)d_in[49];

    float* out = (float*)d_out;
    float* out_hn = out + (size_t)NTOK * ADIM;
    float* out_cn = out_hn + (size_t)BB * HD;

    // ---- workspace bump allocator ----
    char* ws = (char*)d_ws; size_t off = 0;
    auto alloc = [&](size_t bytes) -> char* {
        char* p = ws + off; off += (bytes + 255) & ~(size_t)255; return p;
    };
    _Float16* obsH = (_Float16*)alloc((size_t)NTOK * OBSD * 2);
    _Float16* Rh   = (_Float16*)alloc((size_t)2048 * 512 * 2);
    _Float16* W1h  = (_Float16*)alloc((size_t)512 * 128 * 2);
    _Float16* W2h  = (_Float16*)alloc((size_t)512 * 512 * 2);
    _Float16* Wcgh = (_Float16*)alloc((size_t)512 * 256 * 2);
    _Float16* Wf1h = (_Float16*)alloc((size_t)512 * 256 * 2);
    _Float16* Wf2h = (_Float16*)alloc((size_t)256 * 512 * 2);
    _Float16* Winh = (_Float16*)alloc((size_t)768 * 256 * 2);
    _Float16* Woh  = (_Float16*)alloc((size_t)256 * 256 * 2);
    _Float16* Wxh  = (_Float16*)alloc((size_t)2048 * 512 * 2);
    _Float16* Wph  = (_Float16*)alloc((size_t)32 * 512 * 2);
    _Float16* embb = (_Float16*)alloc((size_t)NTOK * ED * 2);
    float*    wxf  = (float*)alloc((size_t)NTOK * 2048 * 4);
    float*    vgrw = (float*)alloc((size_t)BB * 2048 * 4);
    _Float16* h3h  = (_Float16*)alloc((size_t)NTOK * HD * 2);
    // transformer-stage region
    _Float16* xb   = (_Float16*)alloc(L5N * ED * 2);
    _Float16* lnb  = (_Float16*)alloc(L5N * ED * 2);
    _Float16* qkvb = (_Float16*)alloc(L5N * 768 * 2);  // also reused as FF1 output
    // MLP-stage buffers
    _Float16* h1h  = (_Float16*)alloc((size_t)NTOK * HD * 2);
    _Float16* gab  = (_Float16*)alloc((size_t)NTOK * HD * 2);
    _Float16* gbb  = (_Float16*)alloc((size_t)NTOK * HD * 2);
    _Float16* h2h  = (_Float16*)alloc((size_t)NTOK * HD * 2);
    _Float16* wxh  = (_Float16*)alloc((size_t)NTOK * 2048 * 2);

    auto cvt = [&](const float* s, _Float16* d, long n) {
        cvt_f32_f16<<<dim3((unsigned)((n + 255) / 256)), 256, 0, stream>>>(s, d, n);
    };
    // ---- weight / input conversions to f16 ----
    cvt(obs, obsH, (long)NTOK * OBSD);
    cvt(R,   Rh,   2048L * 512);
    cvt(W1,  W1h,  512L * 128);
    cvt(W2,  W2h,  512L * 512);
    cvt(Wcg, Wcgh, 512L * 256);
    cvt(Wf1, Wf1h, 512L * 256);
    cvt(Wf2, Wf2h, 256L * 512);
    cvt(Win, Winh, 768L * 256);
    cvt(Wo,  Woh,  256L * 256);
    cvt(Wx,  Wxh,  2048L * 512);
    pad_wp<<<dim3((32 * 512 + 255) / 256), 256, 0, stream>>>(Wp, Wph);

    // ---- embeddings -> x (5,N,256) ----
    embed_kernel<<<dim3((NTOK * HALFD + 255) / 256), 256, 0, stream>>>(
        goal, pact, hor, rew, retn,
        cegW, cegb, cege,   // goal
        ceaW, ceab, ceae,   // action
        cehW, cehb, cehe,   // horizon
        cerW, cerb, cere,   // reward
        cetW, cetb, cete,   // ret
        xb);

    const unsigned rows5N = (unsigned)(L5N / 16);          // 10240
    const unsigned rowsN  = (unsigned)(NTOK / 16);         // 2048

    // ---- transformer layer (norm_first, 1 head) ----
    ln_rows<<<dim3((unsigned)((L5N + 7) / 8)), 256, 0, stream>>>(xb, g1, be1, lnb, nullptr, ED, (long)L5N);
    gemm_wmma<256, 0><<<dim3(rows5N, 48), 32, 0, stream>>>(lnb, Winh, bin, nullptr, qkvb, nullptr, 768, 768);
    attn_kernel<<<dim3((NTOK + 7) / 8), 256, 0, stream>>>(qkvb, lnb, NTOK);  // reuse lnb as o
    gemm_wmma<256, 0><<<dim3(rows5N, 16), 32, 0, stream>>>(lnb, Woh, bo, xb, xb, nullptr, ED, ED);
    ln_rows<<<dim3((unsigned)((L5N + 7) / 8)), 256, 0, stream>>>(xb, g2, be2, lnb, nullptr, ED, (long)L5N);
    gemm_wmma<256, 2><<<dim3(rows5N, 32), 32, 0, stream>>>(lnb, Wf1h, bf1, nullptr, qkvb, nullptr, HD, HD);
    gemm_wmma<512, 0><<<dim3(rows5N, 16), 32, 0, stream>>>(qkvb, Wf2h, bf2, xb, xb, nullptr, ED, ED);
    maxL_kernel<<<dim3((NTOK * ED + 255) / 256), 256, 0, stream>>>(xb, embb, NTOK);

    // ---- MLP + gating ----
    gemm_wmma<128, 1><<<dim3(rowsN, 32), 32, 0, stream>>>(obsH, W1h, b1, nullptr, h1h, nullptr, HD, HD);
    gemm_wmma<512, 0><<<dim3(rowsN, 32), 32, 0, stream>>>(h1h, W2h, b2, nullptr, gab, nullptr, HD, HD);
    gemm_wmma<256, 0><<<dim3(rowsN, 32), 32, 0, stream>>>(embb, Wcgh, bcg, nullptr, gbb, nullptr, HD, HD);
    gate_combine<<<dim3((unsigned)(((long)NTOK * HD + 255) / 256)), 256, 0, stream>>>(gab, gbb, h2h, (long)NTOK * HD);

    // ---- wx = LN(h2 @ Wx^T) * gx  (bulk, T-parallel) ----
    gemm_wmma<512, 0><<<dim3(rowsN, 128), 32, 0, stream>>>(h2h, Wxh, nullptr, nullptr, wxh, nullptr, 2048, 2048);
    ln_rows<<<dim3((NTOK + 7) / 8), 256, 0, stream>>>(wxh, gx, nullptr, nullptr, wxf, 2048, (long)NTOK);

    // ---- LSTM scan (single WGP, LDS-resident state) ----
    size_t smem = (size_t)BB * HD * 4 + (size_t)BB * HD * 2 + 2 * (size_t)BB * 16 * 4 + 4 * (size_t)BB * 4;
    hipFuncSetAttribute((const void*)lstm_scan, hipFuncAttributeMaxDynamicSharedMemorySize, (int)smem);
    lstm_scan<<<dim3(1), 1024, smem, stream>>>(Rh, wxf, gr, blstm, gc, bc, h0, c0,
                                               vgrw, h3h, out_hn, out_cn, TT);

    // ---- logits = h3 @ Wp^T + bp ----
    gemm_wmma<512, 0><<<dim3(rowsN, 2), 32, 0, stream>>>(h3h, Wph, bp, nullptr, nullptr, out, ADIM, ADIM);
}
